// SimilarityTreeLSTM_18915035972203
// MI455X (gfx1250) — matrix operand order repaired and checked by reading, not compiled, not run on hardware
//
#include <hip/hip_runtime.h>
#include <math.h>

// ---------------------------------------------------------------------------
// SimilarityTreeLSTM on MI455X (gfx1250, wave32, WMMA, 320KB WGP LDS)
//
// Sizes: N=128 nodes/tree, IN=512, MEM=256 (3*MEM=768), HID=128, NCLS=5
// ---------------------------------------------------------------------------

#define NNODE 128
#define INDIM 512
#define MEMD  256
#define M3    768
#define HIDD  128
#define NCLS  5

typedef __attribute__((ext_vector_type(16))) _Float16 v16h;
typedef __attribute__((ext_vector_type(8)))  _Float16 v8h;
typedef __attribute__((ext_vector_type(8)))  float    v8f;

__device__ __forceinline__ float sigmoidf_(float x) { return 1.0f / (1.0f + expf(-x)); }

// ---------------------------------------------------------------------------
// Prep kernels: gather embedding rows -> f16, transpose weights -> f16 [N][K]
// ---------------------------------------------------------------------------
__global__ void gather_x_kernel(const float* __restrict__ emb,
                                const int* __restrict__ lin,
                                const int* __restrict__ rin,
                                _Float16* __restrict__ Xh) {
  int tid = blockIdx.x * blockDim.x + threadIdx.x;     // 2*128*512 elements
  if (tid >= 2 * NNODE * INDIM) return;
  int k    = tid & (INDIM - 1);
  int i    = (tid >> 9) & (NNODE - 1);
  int tree = tid >> 16;
  const int* in = tree ? rin : lin;
  int tok = in[i];
  Xh[tid] = (_Float16)emb[(size_t)tok * INDIM + k];
}

__global__ void transpose_w_kernel(const float* __restrict__ W,  // [K,N] row-major
                                   _Float16* __restrict__ WT,    // [N,K] row-major
                                   int K, int N) {
  int tid = blockIdx.x * blockDim.x + threadIdx.x;
  if (tid >= K * N) return;
  int k = tid % K;
  int n = tid / K;
  WT[(size_t)n * K + k] = (_Float16)W[(size_t)k * N + n];
}

// ---------------------------------------------------------------------------
// f16 WMMA GEMM: C[M,N] (f32, +bias) = A[M,K] @ B[K,N]
//   A : f16 row-major [M,K];  BT : f16 [N,K] (i.e. B column-major)
// One wave per 16x16 output tile; K stepped by 32 (v_wmma_f32_16x16x32_f16).
// Fragment layouts per CDNA5 ISA 7.12.2 (wave32):
//   A elem e (0..15), lane L: K = (e>>3)*16 + (L>>4)*8 + (e&7), row M = L&15
//   B elem e, lane L:          K = (e>>3)*16 + (L>>4)*8 + (e&7), col N = L&15
//   C elem r, lane L:          row = r + 8*(L>>4),              col = L&15
// ---------------------------------------------------------------------------
__global__ void gemm_wmma_f16(const _Float16* __restrict__ A,
                              const _Float16* __restrict__ BT,
                              const float* __restrict__ bias,
                              float* __restrict__ C,
                              int M, int N, int K) {
  const int wave   = blockIdx.x * (blockDim.x >> 5) + (threadIdx.x >> 5);
  const int tilesN = N >> 4;
  const int tiles  = (M >> 4) * tilesN;
  if (wave >= tiles) return;                      // wave-uniform: EXEC stays all-1s
  const int tm   = wave / tilesN;
  const int tn   = wave % tilesN;
  const int lane = threadIdx.x & 31;
  const int hlf  = lane >> 4;                     // lane half: 0 or 1
  const int rc   = lane & 15;                     // row (A) / col (B,C) within tile

  const _Float16* Arow = A  + (size_t)(tm * 16 + rc) * K;
  const _Float16* Brow = BT + (size_t)(tn * 16 + rc) * K;

  v8f acc = {};
  for (int kb = 0; kb < K; kb += 32) {
    v8h a0 = *(const v8h*)(Arow + kb + hlf * 8);        // K = kb + hlf*8 + 0..7
    v8h a1 = *(const v8h*)(Arow + kb + 16 + hlf * 8);   // K = kb+16 + hlf*8 + 0..7
    v8h b0 = *(const v8h*)(Brow + kb + hlf * 8);        // K = kb + hlf*8 + 0..7
    v8h b1 = *(const v8h*)(Brow + kb + 16 + hlf * 8);   // K = kb+16 + hlf*8 + 0..7
    v16h a = __builtin_shufflevector(a0, a1, 0,1,2,3,4,5,6,7,8,9,10,11,12,13,14,15);
    v16h b = __builtin_shufflevector(b0, b1, 0,1,2,3,4,5,6,7,8,9,10,11,12,13,14,15);
    acc = __builtin_amdgcn_wmma_f32_16x16x32_f16(false, a, false, b,
                                                 (short)0, acc, false, false);
  }

  const int row0 = tm * 16 + hlf * 8;
  const int col  = tn * 16 + rc;
  const float bv = bias[col];
#pragma unroll
  for (int r = 0; r < 8; ++r)
    C[(size_t)(row0 + r) * N + col] = acc[r] + bv;
}

// ---------------------------------------------------------------------------
// Sequential tree scan. One 256-thread workgroup per tree.
// Hs / FCs accumulators ((N+1) x 256 f32 each = 264 KB) live in CDNA5's
// 320 KB-per-WGP LDS (impossible on CDNA4's 64 KB CU) so the 128-step
// dependency chain never leaves the WGP.
// Thread j owns columns {j, 256+j, 512+j} of iou -> gates i/o/u need no
// cross-thread exchange. Weight reads are coalesced in natural [K,N] layout.
// ---------------------------------------------------------------------------
__global__ void tree_scan_kernel(const float* __restrict__ ioux_all, // [2][128][768]
                                 const float* __restrict__ fx_all,   // [2][128][256]
                                 const float* __restrict__ Wiouh,    // [256][768]
                                 const float* __restrict__ biouh,    // [768]
                                 const float* __restrict__ Wfh,      // [256][256]
                                 const float* __restrict__ bfh,      // [256]
                                 const int* __restrict__ lparents,
                                 const int* __restrict__ rparents,
                                 float* __restrict__ H)              // [2][128][256]
{
  const int tree = blockIdx.x;
  const int j    = threadIdx.x;                  // 0..255
  const int* parents  = tree ? rparents : lparents;
  const float* ioux   = ioux_all + (size_t)tree * NNODE * M3;
  const float* fx     = fx_all   + (size_t)tree * NNODE * MEMD;
  float*       Ht     = H        + (size_t)tree * NNODE * MEMD;

  extern __shared__ float smem[];
  float* Hs   = smem;                            // (N+1)*256
  float* FCs  = Hs  + (NNODE + 1) * MEMD;        // (N+1)*256
  float* hcur = FCs + (NNODE + 1) * MEMD;        // 256

  for (int idx = j; idx < 2 * (NNODE + 1) * MEMD; idx += 256) Hs[idx] = 0.0f;
  if (j < MEMD) hcur[j] = 0.0f;
  __syncthreads();

  const float bo0 = biouh[j], bo1 = biouh[MEMD + j], bo2 = biouh[2 * MEMD + j];
  const float bfj = bfh[j];

  for (int t = 0; t < NNODE; ++t) {
    const float* iour = ioux + (size_t)t * M3;
    __builtin_prefetch(iour + M3 + j, 0, 0);     // next node's input row (speculative)

    float a0 = iour[j] + bo0;
    float a1 = iour[MEMD + j] + bo1;
    float a2 = iour[2 * MEMD + j] + bo2;
    const float* hs_t = Hs + t * MEMD;
#pragma unroll 4
    for (int k = 0; k < MEMD; ++k) {
      const float  hv = hs_t[k];                 // LDS broadcast
      const float* wr = Wiouh + (size_t)k * M3;  // coalesced across threads
      a0 = fmaf(hv, wr[j],            a0);
      a1 = fmaf(hv, wr[MEMD + j],     a1);
      a2 = fmaf(hv, wr[2 * MEMD + j], a2);
    }
    const float i_ = sigmoidf_(a0);
    const float o_ = sigmoidf_(a1);
    const float u_ = tanhf(a2);
    const float c  = fmaf(i_, u_, FCs[t * MEMD + j]);
    const float h  = o_ * tanhf(c);
    hcur[j] = h;
    Ht[(size_t)t * MEMD + j] = h;
    const int p = parents[t];                    // p > t; root has p == NNODE (sentinel)
    __syncthreads();                             // hcur complete

    float fp = bfj + (p < NNODE ? fx[(size_t)p * MEMD + j] : 0.0f);
#pragma unroll 4
    for (int k = 0; k < MEMD; ++k)
      fp = fmaf(hcur[k], Wfh[(size_t)k * MEMD + j], fp);
    const float f = sigmoidf_(fp);

    Hs[p * MEMD + j]  += h;                      // thread j owns column j: no conflict
    FCs[p * MEMD + j] += f * c;
    __syncthreads();                             // visible before node t+1 consumes
  }
}

// ---------------------------------------------------------------------------
// Attention + head, algebraically reduced to root rows only:
//   beta_root needs softmax of S[127,:]; alpha_root needs softmax of S[:,127].
// Single 256-thread block.
// ---------------------------------------------------------------------------
__global__ void attn_head_kernel(const float* __restrict__ H,       // [2][128][256]
                                 const float* __restrict__ Wattn,   // [512][256]
                                 const float* __restrict__ battn,   // [256]
                                 const float* __restrict__ Wwh,     // [512][128]
                                 const float* __restrict__ bwh,     // [128]
                                 const float* __restrict__ Wwp,     // [128][5]
                                 const float* __restrict__ bwp,     // [5]
                                 float* __restrict__ out)           // [5]
{
  __shared__ float hlr[MEMD], hrr[MEMD];
  __shared__ float srow[NNODE], scol[NNODE];
  __shared__ float beta[MEMD], alpha[MEMD];
  __shared__ float vl[MEMD], vr[MEMD], vec[2 * MEMD];
  __shared__ float hid[HIDD], red[256], lg[NCLS];
  const int tid = threadIdx.x;
  const float* Hl = H;
  const float* Hr = H + NNODE * MEMD;

  hlr[tid] = Hl[(NNODE - 1) * MEMD + tid];
  hrr[tid] = Hr[(NNODE - 1) * MEMD + tid];
  __syncthreads();

  if (tid < NNODE) {
    float s0 = 0.0f, s1 = 0.0f;
    for (int k = 0; k < MEMD; ++k) {
      s0 = fmaf(Hr[tid * MEMD + k], hlr[k], s0);   // S[127, tid]
      s1 = fmaf(Hl[tid * MEMD + k], hrr[k], s1);   // S[tid, 127]
    }
    srow[tid] = s0; scol[tid] = s1;
  }
  __syncthreads();

  // --- softmax(srow) ---
  red[tid] = (tid < NNODE) ? srow[tid] : -3.0e38f;
  __syncthreads();
  for (int s = 128; s > 0; s >>= 1) { if (tid < s) red[tid] = fmaxf(red[tid], red[tid + s]); __syncthreads(); }
  const float mrow = red[0];
  __syncthreads();
  if (tid < NNODE) srow[tid] = expf(srow[tid] - mrow);
  __syncthreads();
  red[tid] = (tid < NNODE) ? srow[tid] : 0.0f;
  __syncthreads();
  for (int s = 128; s > 0; s >>= 1) { if (tid < s) red[tid] += red[tid + s]; __syncthreads(); }
  const float ssum = red[0];
  __syncthreads();

  // --- softmax(scol) ---
  red[tid] = (tid < NNODE) ? scol[tid] : -3.0e38f;
  __syncthreads();
  for (int s = 128; s > 0; s >>= 1) { if (tid < s) red[tid] = fmaxf(red[tid], red[tid + s]); __syncthreads(); }
  const float mcol = red[0];
  __syncthreads();
  if (tid < NNODE) scol[tid] = expf(scol[tid] - mcol);
  __syncthreads();
  red[tid] = (tid < NNODE) ? scol[tid] : 0.0f;
  __syncthreads();
  for (int s = 128; s > 0; s >>= 1) { if (tid < s) red[tid] += red[tid + s]; __syncthreads(); }
  const float csum = red[0];
  __syncthreads();

  // --- beta_root / alpha_root (weighted sums over nodes) ---
  {
    float b = 0.0f, a = 0.0f;
    for (int i = 0; i < NNODE; ++i) {
      b = fmaf(srow[i], Hr[i * MEMD + tid], b);
      a = fmaf(scol[i], Hl[i * MEMD + tid], a);
    }
    beta[tid]  = b / ssum;
    alpha[tid] = a / csum;
  }
  __syncthreads();

  // --- v_l / v_r (root rows): concat(h_root, ctx) @ Wattn + battn ---
  {
    float sl = battn[tid], sr = battn[tid];
    for (int k = 0; k < MEMD; ++k) {
      const float w0 = Wattn[(size_t)k * MEMD + tid];
      const float w1 = Wattn[(size_t)(MEMD + k) * MEMD + tid];
      sl = fmaf(hlr[k],   w0, sl);  sl = fmaf(beta[k],  w1, sl);
      sr = fmaf(hrr[k],   w0, sr);  sr = fmaf(alpha[k], w1, sr);
    }
    vl[tid] = sl; vr[tid] = sr;
  }
  __syncthreads();

  vec[tid]        = vl[tid] * vr[tid];
  vec[MEMD + tid] = fabsf(vl[tid] - vr[tid]);
  __syncthreads();

  if (tid < HIDD) {
    float s = bwh[tid];
    for (int k = 0; k < 2 * MEMD; ++k) s = fmaf(vec[k], Wwh[(size_t)k * HIDD + tid], s);
    hid[tid] = sigmoidf_(s);
  }
  __syncthreads();

  if (tid < NCLS) {
    float s = bwp[tid];
    for (int jj = 0; jj < HIDD; ++jj) s = fmaf(hid[jj], Wwp[jj * NCLS + tid], s);
    lg[tid] = s;
  }
  __syncthreads();

  if (tid == 0) {
    float m = lg[0];
    for (int c = 1; c < NCLS; ++c) m = fmaxf(m, lg[c]);
    float sum = 0.0f;
    for (int c = 0; c < NCLS; ++c) sum += expf(lg[c] - m);
    const float lse = m + logf(sum);
    for (int c = 0; c < NCLS; ++c) out[c] = lg[c] - lse;
  }
}

// ---------------------------------------------------------------------------
// Host launcher
// ---------------------------------------------------------------------------
extern "C" void kernel_launch(void* const* d_in, const int* in_sizes, int n_in,
                              void* d_out, int out_size, void* d_ws, size_t ws_size,
                              hipStream_t stream) {
  (void)in_sizes; (void)n_in; (void)out_size; (void)ws_size;

  const float* emb      = (const float*)d_in[0];
  const float* Wioux    = (const float*)d_in[1];
  const float* bioux    = (const float*)d_in[2];
  const float* Wiouh    = (const float*)d_in[3];
  const float* biouh    = (const float*)d_in[4];
  const float* Wfx      = (const float*)d_in[5];
  const float* bfx      = (const float*)d_in[6];
  const float* Wfh      = (const float*)d_in[7];
  const float* bfh      = (const float*)d_in[8];
  const float* Wattn    = (const float*)d_in[9];
  const float* battn    = (const float*)d_in[10];
  const float* Wwh      = (const float*)d_in[11];
  const float* bwh      = (const float*)d_in[12];
  const float* Wwp      = (const float*)d_in[13];
  const float* bwp      = (const float*)d_in[14];
  const int*   linputs  = (const int*)d_in[15];
  const int*   lparents = (const int*)d_in[16];
  const int*   rinputs  = (const int*)d_in[17];
  const int*   rparents = (const int*)d_in[18];
  float* out = (float*)d_out;

  // Workspace layout (bytes)
  char* ws = (char*)d_ws;
  _Float16* Xh       = (_Float16*)(ws + 0);            // 2*128*512 f16  = 256 KB
  _Float16* WiouxT_h = (_Float16*)(ws + 262144);       // 768*512  f16   = 768 KB
  _Float16* WfxT_h   = (_Float16*)(ws + 1048576);      // 256*512  f16   = 256 KB
  float*    ioux_all = (float*)   (ws + 1310720);      // 2*128*768 f32  = 768 KB
  float*    fx_all   = (float*)   (ws + 2097152);      // 2*128*256 f32  = 256 KB
  float*    Hbuf     = (float*)   (ws + 2359296);      // 2*128*256 f32  = 256 KB
                                                       // total          ~2.56 MB

  // --- Phase 0: gather + f16 conversion / weight transposes ---
  gather_x_kernel<<<(2 * NNODE * INDIM) / 256, 256, 0, stream>>>(emb, linputs, rinputs, Xh);
  transpose_w_kernel<<<(INDIM * M3) / 256,   256, 0, stream>>>(Wioux, WiouxT_h, INDIM, M3);
  transpose_w_kernel<<<(INDIM * MEMD) / 256, 256, 0, stream>>>(Wfx,   WfxT_h,   INDIM, MEMD);

  // --- Phase 1: WMMA input projections (per tree) ---
  // tiles(iou) = 8*48 = 384 -> 96 blocks of 4 waves; tiles(fx) = 8*16 = 128 -> 32 blocks
  for (int tree = 0; tree < 2; ++tree) {
    const _Float16* A = Xh + (size_t)tree * NNODE * INDIM;
    gemm_wmma_f16<<<96, 128, 0, stream>>>(A, WiouxT_h, bioux,
                                          ioux_all + (size_t)tree * NNODE * M3,
                                          NNODE, M3, INDIM);
    gemm_wmma_f16<<<32, 128, 0, stream>>>(A, WfxT_h, bfx,
                                          fx_all + (size_t)tree * NNODE * MEMD,
                                          NNODE, MEMD, INDIM);
  }

  // --- Phase 2: sequential tree scan, one WGP per tree, 264 KB LDS state ---
  const size_t scan_lds = (size_t)(2 * (NNODE + 1) * MEMD + MEMD) * sizeof(float); // 265,216 B
  hipFuncSetAttribute((const void*)tree_scan_kernel,
                      hipFuncAttributeMaxDynamicSharedMemorySize, (int)scan_lds);
  tree_scan_kernel<<<2, 256, scan_lds, stream>>>(ioux_all, fx_all, Wiouh, biouh,
                                                 Wfh, bfh, lparents, rparents, Hbuf);

  // --- Phase 3: root-only attention + classification head ---
  attn_head_kernel<<<1, 256, 0, stream>>>(Hbuf, Wattn, battn, Wwh, bwh, Wwp, bwp, out);
}
